// StyleDecorator_36721970381148
// MI455X (gfx1250) — matrix operand order repaired and checked by reading, not compiled, not run on hardware
//
#include <hip/hip_runtime.h>

// ---------------------------------------------------------------------------
// StyleDecorator pipeline for MI455X (gfx1250, wave32).
//  - 309-GFLOP patch-correlation GEMM: bf16 V_WMMA_F32_16X16X32_BF16, LDS
//    double-buffered via GLOBAL_LOAD_ASYNC_TO_LDS_B128 (ASYNCcnt), fused
//    per-column argmax (packed u64 atomic max).
//  - eigh replaced by coupled Newton-Schulz; all fp32 256^3-class GEMMs run
//    on V_WMMA_F32_16X16X4_F32 (full fp32 matrix pipe).
//  - per-batch bf16 patch matrices (2x18.9MB) stay resident in the 192MB L2.
// ---------------------------------------------------------------------------

#define B_  4
#define C_  256
#define N_  4096          // 64*64 pixels
#define P_  4096          // patches == pixels
#define CQ_ 2304          // C_*9
#define CC_ 65536         // C_*C_
#define NS_ITERS 16

#define USE_ASYNC_LDS 1   // set 0 to fall back to sync ds-store tile loading

typedef __attribute__((ext_vector_type(16))) __bf16 v16bf;
typedef __attribute__((ext_vector_type(8)))  __bf16 v8bf;
typedef __attribute__((ext_vector_type(8)))  float  v8f;
typedef __attribute__((ext_vector_type(2)))  float  v2f;

__device__ __forceinline__ __bf16 f2bf(float f) {
  unsigned u = __float_as_uint(f);
  u = (u + 0x7fffu + ((u >> 16) & 1u)) >> 16;   // round-to-nearest-even
  unsigned short s = (unsigned short)u;
  return __builtin_bit_cast(__bf16, s);
}

__device__ __forceinline__ unsigned monoKey(float f) {
  unsigned u = __float_as_uint(f);
  return (u & 0x80000000u) ? ~u : (u | 0x80000000u);   // order-preserving f32->u32
}

// --------------------------- means -----------------------------------------
__global__ void __launch_bounds__(256)
mean_k(const float* __restrict__ content, const float* __restrict__ style,
       float* __restrict__ meanC, float* __restrict__ meanS) {
  __shared__ float red[256];
  int blk = blockIdx.x;                 // 0..2047
  int f = blk >> 10, bc = blk & 1023;   // bc = b*256 + c
  const float* src = (f ? style : content) + (size_t)bc * N_;
  float s = 0.f;
  for (int i = threadIdx.x; i < N_; i += 256) s += src[i];
  red[threadIdx.x] = s; __syncthreads();
  for (int w = 128; w > 0; w >>= 1) {
    if (threadIdx.x < w) red[threadIdx.x] += red[threadIdx.x + w];
    __syncthreads();
  }
  if (threadIdx.x == 0) (f ? meanS : meanC)[bc] = red[0] * (1.0f / (float)N_);
}

// --------------------------- covariance (scatter), f32 WMMA ----------------
// Cov[i][j] = sum_n (X[i][n]-m_i)(X[j][n]-m_j).  grid (4,4,8), 256 thr.
// 8 waves as 2x4; wave tile 32(i) x 16(j); K = pixels in chunks of 16.
__global__ void __launch_bounds__(256)
cov_k(const float* __restrict__ content, const float* __restrict__ style,
      const float* __restrict__ meanC, const float* __restrict__ meanS,
      float* __restrict__ covC, float* __restrict__ covS) {
  __shared__ float As[64][17];
  __shared__ float Bs[16][65];
  int pr = blockIdx.z; int f = pr >> 2; int b = pr & 3;
  const float* X   = (f ? style : content) + (size_t)b * C_ * N_;
  const float* mn  = (f ? meanS : meanC) + b * C_;
  float*       Cov = (f ? covS : covC) + (size_t)b * CC_;
  const int tid = threadIdx.x;
  const int lane = tid & 31, wave = tid >> 5;
  const int wr = wave >> 2, wcl = wave & 3;
  const int lr = lane & 15, lh = lane >> 4;
  const int row0 = blockIdx.x * 64, col0 = blockIdx.y * 64;
  const int r = tid >> 2, kk4 = (tid & 3) * 4;
  float ma = mn[row0 + r];
  float mb = mn[col0 + r];
  v8f acc[2];
  const v8f vz = {0.f,0.f,0.f,0.f,0.f,0.f,0.f,0.f};
  acc[0] = vz; acc[1] = vz;
  for (int n0 = 0; n0 < N_; n0 += 16) {
#pragma unroll
    for (int u = 0; u < 4; ++u) As[r][kk4 + u] = X[(size_t)(row0 + r) * N_ + n0 + kk4 + u] - ma;
#pragma unroll
    for (int u = 0; u < 4; ++u) Bs[kk4 + u][r] = X[(size_t)(col0 + r) * N_ + n0 + kk4 + u] - mb;
    __syncthreads();
#pragma unroll
    for (int ks = 0; ks < 16; ks += 4) {
      v2f bfrag = { Bs[ks + lh * 2][wcl * 16 + lr], Bs[ks + lh * 2 + 1][wcl * 16 + lr] };
#pragma unroll
      for (int i = 0; i < 2; ++i) {
        int arow = wr * 32 + i * 16 + lr;
        v2f afrag = { As[arow][ks + lh * 2], As[arow][ks + lh * 2 + 1] };
        acc[i] = __builtin_amdgcn_wmma_f32_16x16x4_f32(false, afrag, false, bfrag,
                                                       (short)0, acc[i], false, false);
      }
    }
    __syncthreads();
  }
#pragma unroll
  for (int i = 0; i < 2; ++i)
#pragma unroll
    for (int rr = 0; rr < 8; ++rr) {
      int gr = row0 + wr * 32 + i * 16 + lh * 8 + rr;
      int gc = col0 + wcl * 16 + lr;
      Cov[gr * C_ + gc] = acc[i][rr];
    }
}

// --------------------------- Frobenius norms -------------------------------
__global__ void __launch_bounds__(256)
frob_k(const float* __restrict__ covC, const float* __restrict__ covS,
       float* __restrict__ norms) {
  __shared__ float red[256];
  int pr = blockIdx.x;
  const float* cov = (pr < 4) ? covC + (size_t)pr * CC_ : covS + (size_t)(pr - 4) * CC_;
  float s = 0.f;
  for (int i = threadIdx.x; i < CC_; i += 256) { float v = cov[i]; s += v * v; }
  red[threadIdx.x] = s; __syncthreads();
  for (int w = 128; w > 0; w >>= 1) {
    if (threadIdx.x < w) red[threadIdx.x] += red[threadIdx.x + w];
    __syncthreads();
  }
  if (threadIdx.x == 0) norms[pr] = sqrtf(red[0]);
}

// --------------------------- Newton-Schulz init ----------------------------
__global__ void __launch_bounds__(256)
init_ns_k(const float* __restrict__ covC, const float* __restrict__ covS,
          const float* __restrict__ norms, float* __restrict__ Y, float* __restrict__ Z) {
  int pr = blockIdx.y;
  int i = blockIdx.x * 256 + threadIdx.x;     // 0..65535
  const float* cov = (pr < 4) ? covC + (size_t)pr * CC_ : covS + (size_t)(pr - 4) * CC_;
  float inv = 1.0f / norms[pr];
  Y[(size_t)pr * CC_ + i] = cov[i] * inv;
  Z[(size_t)pr * CC_ + i] = ((i >> 8) == (i & 255)) ? 1.0f : 0.0f;
}

// C = aI*I + beta*(A @ B), 256x256x256, grid (4,4,8), f32 WMMA.
__global__ void __launch_bounds__(256)
ns_gemm_k(const float* __restrict__ A, const float* __restrict__ Bm,
          float* __restrict__ Cm, float aI, float beta) {
  __shared__ float As[64][17];
  __shared__ float Bs[16][65];
  int pr = blockIdx.z;
  const float* Ap = A  + (size_t)pr * CC_;
  const float* Bp = Bm + (size_t)pr * CC_;
  float*       Cp = Cm + (size_t)pr * CC_;
  const int tid = threadIdx.x;
  const int lane = tid & 31, wave = tid >> 5;
  const int wr = wave >> 2, wcl = wave & 3;        // 2x4 waves; wave tile 32x16
  const int lr = lane & 15, lh = lane >> 4;
  const int row0 = blockIdx.x * 64, col0 = blockIdx.y * 64;
  const int r = tid >> 2, kk4 = (tid & 3) * 4;
  const int rb = tid >> 4, cb = (tid & 15) * 4;
  v8f acc[2];
  const v8f vz = {0.f,0.f,0.f,0.f,0.f,0.f,0.f,0.f};
  acc[0] = vz; acc[1] = vz;
  for (int k0 = 0; k0 < C_; k0 += 16) {
#pragma unroll
    for (int u = 0; u < 4; ++u) As[r][kk4 + u] = Ap[(row0 + r) * C_ + k0 + kk4 + u];
#pragma unroll
    for (int u = 0; u < 4; ++u) Bs[rb][cb + u] = Bp[(k0 + rb) * C_ + col0 + cb + u];
    __syncthreads();
#pragma unroll
    for (int ks = 0; ks < 16; ks += 4) {
      v2f bfrag = { Bs[ks + lh * 2][wcl * 16 + lr], Bs[ks + lh * 2 + 1][wcl * 16 + lr] };
#pragma unroll
      for (int i = 0; i < 2; ++i) {
        int arow = wr * 32 + i * 16 + lr;
        v2f afrag = { As[arow][ks + lh * 2], As[arow][ks + lh * 2 + 1] };
        acc[i] = __builtin_amdgcn_wmma_f32_16x16x4_f32(false, afrag, false, bfrag,
                                                       (short)0, acc[i], false, false);
      }
    }
    __syncthreads();
  }
#pragma unroll
  for (int i = 0; i < 2; ++i)
#pragma unroll
    for (int rr = 0; rr < 8; ++rr) {
      int gr = row0 + wr * 32 + i * 16 + lh * 8 + rr;
      int gc = col0 + wcl * 16 + lr;
      Cp[gr * C_ + gc] = beta * acc[i][rr] + ((gr == gc) ? aI : 0.0f);
    }
}

// de-normalize NS results into the final transform matrices
__global__ void __launch_bounds__(256)
scale_k(const float* __restrict__ Y, const float* __restrict__ Z,
        const float* __restrict__ norms,
        float* __restrict__ MinvC, float* __restrict__ MinvS, float* __restrict__ MsqS) {
  int b = blockIdx.y;
  int i = blockIdx.x * 256 + threadIdx.x;
  float rc = rsqrtf(norms[b]);
  float rs = rsqrtf(norms[4 + b]);
  float ss = sqrtf(norms[4 + b]);
  MinvC[(size_t)b * CC_ + i] = Z[(size_t)b * CC_ + i] * rc;
  MinvS[(size_t)b * CC_ + i] = Z[(size_t)(4 + b) * CC_ + i] * rs;
  MsqS [(size_t)b * CC_ + i] = Y[(size_t)(4 + b) * CC_ + i] * ss;
}

// --------------------------- apply transform, f32 WMMA ---------------------
// Out[b][c][n] = sum_d M[b][c][d]*(X[b][d][n]-meanSub[b][d]) + meanAdd[b][c]
// grid (4, 64, 4), 256 thr; 8 waves 2x4; wave tile 32(c) x 16(n).
__global__ void __launch_bounds__(256)
apply_gemm_k(const float* __restrict__ M, const float* __restrict__ X,
             const float* __restrict__ meanSub, const float* __restrict__ meanAdd,
             float* __restrict__ Out) {
  __shared__ float Ms[64][17];
  __shared__ float Xs[16][65];
  int b = blockIdx.z;
  const int tid = threadIdx.x;
  const int lane = tid & 31, wave = tid >> 5;
  const int wr = wave >> 2, wcl = wave & 3;
  const int lr = lane & 15, lh = lane >> 4;
  const int row0 = blockIdx.x * 64, col0 = blockIdx.y * 64;
  const float* Mp = M + (size_t)b * CC_;
  const float* Xp = X + (size_t)b * C_ * N_;
  const int r = tid >> 2, kk4 = (tid & 3) * 4;
  const int rb = tid >> 4, cb = (tid & 15) * 4;
  v8f acc[2];
  const v8f vz = {0.f,0.f,0.f,0.f,0.f,0.f,0.f,0.f};
  acc[0] = vz; acc[1] = vz;
  for (int k0 = 0; k0 < C_; k0 += 16) {
#pragma unroll
    for (int u = 0; u < 4; ++u) Ms[r][kk4 + u] = Mp[(row0 + r) * C_ + k0 + kk4 + u];
    float msub = meanSub ? meanSub[b * C_ + k0 + rb] : 0.0f;
#pragma unroll
    for (int u = 0; u < 4; ++u)
      Xs[rb][cb + u] = Xp[(size_t)(k0 + rb) * N_ + col0 + cb + u] - msub;
    __syncthreads();
#pragma unroll
    for (int ks = 0; ks < 16; ks += 4) {
      v2f bfrag = { Xs[ks + lh * 2][wcl * 16 + lr], Xs[ks + lh * 2 + 1][wcl * 16 + lr] };
#pragma unroll
      for (int i = 0; i < 2; ++i) {
        int arow = wr * 32 + i * 16 + lr;
        v2f afrag = { Ms[arow][ks + lh * 2], Ms[arow][ks + lh * 2 + 1] };
        acc[i] = __builtin_amdgcn_wmma_f32_16x16x4_f32(false, afrag, false, bfrag,
                                                       (short)0, acc[i], false, false);
      }
    }
    __syncthreads();
  }
#pragma unroll
  for (int i = 0; i < 2; ++i)
#pragma unroll
    for (int rr = 0; rr < 8; ++rr) {
      int gr = row0 + wr * 32 + i * 16 + lh * 8 + rr;
      int gc = col0 + wcl * 16 + lr;
      float madd = meanAdd ? meanAdd[b * C_ + gr] : 0.0f;
      Out[(size_t)(b * C_ + gr) * N_ + gc] = acc[i][rr] + madd;
    }
}

// --------------------------- patch norms -----------------------------------
__global__ void __launch_bounds__(256)
s2_k(const float* __restrict__ wsf, float* __restrict__ s2) {
  int b = blockIdx.x >> 4;
  int n = (blockIdx.x & 15) * 256 + threadIdx.x;
  float s = 0.f;
  for (int c = 0; c < C_; ++c) {
    float v = wsf[((size_t)(b * C_ + c) << 12) + n];
    s += v * v;
  }
  s2[b * N_ + n] = s;
}

__global__ void __launch_bounds__(256)
kinv_k(const float* __restrict__ s2, float* __restrict__ kinv) {
  int b = blockIdx.x >> 4;
  int n = (blockIdx.x & 15) * 256 + threadIdx.x;
  int y = n >> 6, x = n & 63;
  float s = 0.f;
#pragma unroll
  for (int di = 0; di < 3; ++di)
#pragma unroll
    for (int dj = 0; dj < 3; ++dj) {
      int sy = y + di - 1, sx = x + dj - 1;
      if ((unsigned)sy < 64u && (unsigned)sx < 64u) s += s2[b * N_ + sy * 64 + sx];
    }
  kinv[b * N_ + n] = rsqrtf(s);
}

// --------------------------- bf16 patch matrices ---------------------------
__global__ void __launch_bounds__(256)
build_kn_k(const float* __restrict__ wsf, const float* __restrict__ kinv,
           __bf16* __restrict__ KN, int b) {
  int p = blockIdx.x, c = threadIdx.x;
  int pi = p >> 6, pj = p & 63;
  float s = kinv[b * N_ + p];
  const float* base = wsf + ((size_t)(b * C_ + c)) * N_;
  __bf16* o = KN + (size_t)p * CQ_ + c * 9;
#pragma unroll
  for (int di = 0; di < 3; ++di)
#pragma unroll
    for (int dj = 0; dj < 3; ++dj) {
      int sy = pi + di - 1, sx = pj + dj - 1;
      float v = ((unsigned)sy < 64u && (unsigned)sx < 64u) ? base[sy * 64 + sx] * s : 0.f;
      o[di * 3 + dj] = f2bf(v);
    }
}

__global__ void __launch_bounds__(256)
build_cpt_k(const float* __restrict__ wc, __bf16* __restrict__ CPt, int b) {
  int n = blockIdx.x, c = threadIdx.x;
  int y = n >> 6, x = n & 63;
  const float* base = wc + ((size_t)(b * C_ + c)) * N_;
  __bf16* o = CPt + (size_t)n * CQ_ + c * 9;
#pragma unroll
  for (int di = 0; di < 3; ++di)
#pragma unroll
    for (int dj = 0; dj < 3; ++dj) {
      int sy = y + di - 1, sx = x + dj - 1;
      float v = ((unsigned)sy < 64u && (unsigned)sx < 64u) ? base[sy * 64 + sx] : 0.f;
      o[di * 3 + dj] = f2bf(v);
    }
}

__global__ void __launch_bounds__(256)
amax_init_k(unsigned long long* __restrict__ amax) {
  amax[blockIdx.x * 256 + threadIdx.x] = 0ull;
}

// --------------------------- the big WMMA GEMM + fused argmax --------------
// score[p][n] = sum_cq KN[p][cq] * CPt[n][cq]; argmax over p per column n.
// grid (32 n-blocks, 8 p-blocks), 256 threads (8 waves). Each WG: 512p x 128n.
// LDS tiles double-buffered; filled with GLOBAL_LOAD_ASYNC_TO_LDS_B128
// (ASYNCcnt) so the next tile's DMA overlaps the current tile's WMMAs.
__global__ void __launch_bounds__(256)
score_k(const __bf16* __restrict__ KN, const __bf16* __restrict__ CPt,
        unsigned long long* __restrict__ amax, int b) {
  __shared__ __bf16 At[2][128][40];    // p-rows x 32k (+8 pad), 16B-aligned rows
  __shared__ __bf16 Bt[2][128][40];    // n-rows x 32k
  __shared__ unsigned long long best[128];

  const int tid  = threadIdx.x;
  const int lane = tid & 31, wave = tid >> 5;
  const int wp = wave >> 2, wn = wave & 3;      // 2x4 wave grid
  const int lr = lane & 15, lh = lane >> 4;     // lane row/col, lane half
  const int nBase = blockIdx.x * 128;
  const int pWG   = blockIdx.y * 512;

  if (tid < 128) best[tid] = 0ull;

  const int ldr = tid >> 1;            // 0..127
  const int ldk = (tid & 1) * 16;      // 0 or 16
  const __bf16* gB = CPt + (size_t)(nBase + ldr) * CQ_ + ldk;

  for (int pt = 0; pt < 4; ++pt) {
    const int pBase = pWG + pt * 128;
    const __bf16* gA = KN + (size_t)(pBase + ldr) * CQ_ + ldk;
    v8f acc[4][2];
    const v8f vz = {0.f, 0.f, 0.f, 0.f, 0.f, 0.f, 0.f, 0.f};
#pragma unroll
    for (int i = 0; i < 4; ++i)
#pragma unroll
      for (int j = 0; j < 2; ++j) acc[i][j] = vz;

    __syncthreads();   // all waves done reading buffers / best init visible

#if USE_ASYNC_LDS
    {
      unsigned la = (unsigned)(unsigned long long)(const void*)&At[0][ldr][ldk];
      unsigned lb = (unsigned)(unsigned long long)(const void*)&Bt[0][ldr][ldk];
      asm volatile("global_load_async_to_lds_b128 %0, %1, off"
                   :: "v"(la), "v"(gA) : "memory");
      asm volatile("global_load_async_to_lds_b128 %0, %1, off"
                   :: "v"(lb), "v"(gB) : "memory");
    }
#endif

    for (int k0 = 0; k0 < CQ_; k0 += 32) {
      const int cur = (k0 >> 5) & 1;
#if USE_ASYNC_LDS
      asm volatile("s_wait_asynccnt 0x0" ::: "memory");
      __syncthreads();                 // everyone's current tile is in LDS
      if (k0 + 32 < CQ_) {             // prefetch next tile (overlaps WMMAs)
        unsigned la = (unsigned)(unsigned long long)(const void*)&At[cur ^ 1][ldr][ldk];
        unsigned lb = (unsigned)(unsigned long long)(const void*)&Bt[cur ^ 1][ldr][ldk];
        asm volatile("global_load_async_to_lds_b128 %0, %1, off"
                     :: "v"(la), "v"(gA + k0 + 32) : "memory");
        asm volatile("global_load_async_to_lds_b128 %0, %1, off"
                     :: "v"(lb), "v"(gB + k0 + 32) : "memory");
      }
#else
      *(uint4*)(&At[cur][ldr][ldk]) = *(const uint4*)(gA + k0);
      *(uint4*)(&Bt[cur][ldr][ldk]) = *(const uint4*)(gB + k0);
      __syncthreads();
#endif

      v16bf fa[4], fb[2];
      const int oa = lh ? 8 : 0;       // A: K 0-7/16-23 (lo lanes), 8-15/24-31 (hi)
#pragma unroll
      for (int i = 0; i < 4; ++i) {
        int row = wp * 64 + i * 16 + lr;
        v8bf lo = *(const v8bf*)(&At[cur][row][oa]);
        v8bf hi = *(const v8bf*)(&At[cur][row][oa + 16]);
#pragma unroll
        for (int e = 0; e < 8; ++e) { fa[i][e] = lo[e]; fa[i][e + 8] = hi[e]; }
      }
      const int ob = lh ? 8 : 0;       // B: K 0-7/16-23 (lo lanes), 8-15/24-31 (hi)
#pragma unroll
      for (int j = 0; j < 2; ++j) {
        int row = wn * 32 + j * 16 + lr;
        v8bf lo = *(const v8bf*)(&Bt[cur][row][ob]);
        v8bf hi = *(const v8bf*)(&Bt[cur][row][ob + 16]);
#pragma unroll
        for (int e = 0; e < 8; ++e) { fb[j][e] = lo[e]; fb[j][e + 8] = hi[e]; }
      }
#pragma unroll
      for (int i = 0; i < 4; ++i)
#pragma unroll
        for (int j = 0; j < 2; ++j)
          acc[i][j] = __builtin_amdgcn_wmma_f32_16x16x32_bf16(
              false, fa[i], false, fb[j], (short)0, acc[i][j], false, false);
#if !USE_ASYNC_LDS
      __syncthreads();
#endif
    }

    // fused argmax: per lane, each j owns one column (n = wn*32+j*16+lr)
#pragma unroll
    for (int j = 0; j < 2; ++j) {
      float bv = -3.402823466e38f; int bm = 0;
#pragma unroll
      for (int i = 0; i < 4; ++i)
#pragma unroll
        for (int rr = 0; rr < 8; ++rr) {
          float v = acc[i][j][rr];
          int m = wp * 64 + i * 16 + lh * 8 + rr;  // ascending -> ties keep smallest p
          if (v > bv) { bv = v; bm = m; }
        }
      unsigned p = (unsigned)(pBase + bm);
      unsigned long long key =
          ((unsigned long long)monoKey(bv) << 32) | (unsigned long long)(4095u - p);
      unsigned klo = (unsigned)key, khi = (unsigned)(key >> 32);
      unsigned olo = __shfl_xor(klo, 16, 32);
      unsigned ohi = __shfl_xor(khi, 16, 32);
      unsigned long long ok = ((unsigned long long)ohi << 32) | (unsigned long long)olo;
      if (ok > key) key = ok;
      if (lh == 0) atomicMax(&best[wn * 32 + j * 16 + lr], key);
    }
  }

  __syncthreads();
  if (tid < 128) atomicMax(&amax[(size_t)b * N_ + nBase + tid], best[tid]);
}

// --------------------------- overlap-add reassembly ------------------------
__global__ void __launch_bounds__(256)
reassemble_k(const float* __restrict__ wsf, const unsigned long long* __restrict__ amax,
             float* __restrict__ reass) {
  int pix = blockIdx.x & 4095;
  int b   = blockIdx.x >> 12;
  int y = pix >> 6, x = pix & 63;
  int c = threadIdx.x;
  const float* base = wsf + ((size_t)(b * C_ + c) << 12);
  float acc = 0.f;
#pragma unroll
  for (int di = 0; di < 3; ++di)
#pragma unroll
    for (int dj = 0; dj < 3; ++dj) {
      int h0 = y + 1 - di, w0 = x + 1 - dj;
      if ((unsigned)h0 < 64u && (unsigned)w0 < 64u) {
        unsigned long long key = amax[(size_t)b * N_ + h0 * 64 + w0];
        int p = 4095 - (int)(key & 0xffffffffull);
        int pi = p >> 6, pj = p & 63;
        int sy = pi + di - 1, sx = pj + dj - 1;
        if ((unsigned)sy < 64u && (unsigned)sx < 64u) acc += base[sy * 64 + sx];
      }
    }
  float fy = (y == 0 || y == 63) ? 2.f : 3.f;
  float fx = (x == 0 || x == 63) ? 2.f : 3.f;
  reass[((size_t)(b * C_ + c) << 12) + pix] = acc / (fy * fx);
}

// ===========================================================================
extern "C" void kernel_launch(void* const* d_in, const int* in_sizes, int n_in,
                              void* d_out, int out_size, void* d_ws, size_t ws_size,
                              hipStream_t stream) {
  (void)in_sizes; (void)n_in; (void)out_size; (void)ws_size;
  const float* content = (const float*)d_in[0];
  const float* style   = (const float*)d_in[1];
  float* out = (float*)d_out;

  char* w = (char*)d_ws;
  auto carve = [&](size_t bytes) -> void* {
    void* p = (void*)w;
    w += (bytes + 255) & ~(size_t)255;
    return p;
  };
  float* meanC = (float*)carve(B_ * C_ * 4);
  float* meanS = (float*)carve(B_ * C_ * 4);
  float* norms = (float*)carve(8 * 4);
  float* covC  = (float*)carve((size_t)B_ * CC_ * 4);
  float* covS  = (float*)carve((size_t)B_ * CC_ * 4);
  float* nsY   = (float*)carve((size_t)8 * CC_ * 4);
  float* nsZ   = (float*)carve((size_t)8 * CC_ * 4);
  float* nsY2  = (float*)carve((size_t)8 * CC_ * 4);
  float* nsZ2  = (float*)carve((size_t)8 * CC_ * 4);
  float* nsT   = (float*)carve((size_t)8 * CC_ * 4);
  float* MinvC = (float*)carve((size_t)B_ * CC_ * 4);
  float* MinvS = (float*)carve((size_t)B_ * CC_ * 4);
  float* MsqS  = (float*)carve((size_t)B_ * CC_ * 4);
  float* wc    = (float*)carve((size_t)B_ * C_ * N_ * 4);
  float* wsf   = (float*)carve((size_t)B_ * C_ * N_ * 4);
  float* reass = (float*)carve((size_t)B_ * C_ * N_ * 4);
  float* s2    = (float*)carve((size_t)B_ * N_ * 4);
  float* kinv  = (float*)carve((size_t)B_ * N_ * 4);
  unsigned long long* amax = (unsigned long long*)carve((size_t)B_ * N_ * 8);
  __bf16* KN  = (__bf16*)carve((size_t)P_ * CQ_ * 2);   // per-batch, reused
  __bf16* CPt = (__bf16*)carve((size_t)N_ * CQ_ * 2);   // per-batch, reused

  // 1) stats
  mean_k<<<2048, 256, 0, stream>>>(content, style, meanC, meanS);
  cov_k<<<dim3(4, 4, 8), 256, 0, stream>>>(content, style, meanC, meanS, covC, covS);
  frob_k<<<8, 256, 0, stream>>>(covC, covS, norms);

  // 2) Newton-Schulz: Y -> A^{1/2}, Z -> A^{-1/2} (A = cov/||cov||_F)
  init_ns_k<<<dim3(256, 8), 256, 0, stream>>>(covC, covS, norms, nsY, nsZ);
  float *yc = nsY, *zc = nsZ, *yn = nsY2, *zn = nsZ2;
  for (int it = 0; it < NS_ITERS; ++it) {
    ns_gemm_k<<<dim3(4, 4, 8), 256, 0, stream>>>(zc, yc, nsT, 1.5f, -0.5f); // T = 1.5I - 0.5 Z Y
    ns_gemm_k<<<dim3(4, 4, 8), 256, 0, stream>>>(yc, nsT, yn, 0.0f, 1.0f);  // Y' = Y T
    ns_gemm_k<<<dim3(4, 4, 8), 256, 0, stream>>>(nsT, zc, zn, 0.0f, 1.0f);  // Z' = T Z
    float* t;
    t = yc; yc = yn; yn = t;
    t = zc; zc = zn; zn = t;
  }
  scale_k<<<dim3(256, 4), 256, 0, stream>>>(yc, zc, norms, MinvC, MinvS, MsqS);

  // 3) whitening
  apply_gemm_k<<<dim3(4, 64, 4), 256, 0, stream>>>(MinvC, content, meanC, nullptr, wc);
  apply_gemm_k<<<dim3(4, 64, 4), 256, 0, stream>>>(MinvS, style,   meanS, nullptr, wsf);

  // 4) patch norms
  s2_k  <<<64, 256, 0, stream>>>(wsf, s2);
  kinv_k<<<64, 256, 0, stream>>>(s2, kinv);

  // 5) per-batch: bf16 patch matrices + WMMA score GEMM with fused argmax
  amax_init_k<<<64, 256, 0, stream>>>(amax);
  for (int b = 0; b < B_; ++b) {
    build_kn_k <<<4096, 256, 0, stream>>>(wsf, kinv, KN, b);
    build_cpt_k<<<4096, 256, 0, stream>>>(wc, CPt, b);
    score_k<<<dim3(32, 8), 256, 0, stream>>>(KN, CPt, amax, b);
  }

  // 6) overlap-add reassembly + deconv normalization
  reassemble_k<<<16384, 256, 0, stream>>>(wsf, amax, reass);

  // 7) coloring (style_strength == 1.0 -> output is the stylized feature)
  apply_gemm_k<<<dim3(4, 64, 4), 256, 0, stream>>>(MsqS, reass, nullptr, meanS, out);
}
